// MambaBlock_54520314856000
// MI455X (gfx1250) — matrix operand order repaired and checked by reading, not compiled, not run on hardware
//
#include <hip/hip_runtime.h>
#include <hip/hip_bf16.h>

// Problem constants (from reference)
#define DM 512
#define DS 16
#define DC 4
#define DI 1024        // D_INNER
#define BB 2
#define LL 2048
#define NROWS (BB * LL)   // 4096

typedef __attribute__((ext_vector_type(16))) __bf16 v16bf;
typedef __attribute__((ext_vector_type(8)))  float  v8f;
typedef __attribute__((ext_vector_type(4)))  int    v4i;

// CDNA5 async global->LDS copy path (guarded; host pass and older toolchains
// fall back to a VGPR-staged copy with identical LDS layout).
// Builtin signature: (v4i AS1*, v4i AS3*, Imm offset, Imm cpol)
#if defined(__AMDGCN__) && __has_builtin(__builtin_amdgcn_global_load_async_to_lds_b128)
#define HAVE_ASYNC_LDS 1
typedef __attribute__((address_space(1))) v4i gas_v4i;
typedef __attribute__((address_space(3))) v4i las_v4i;
#endif

__device__ __forceinline__ void copy16_to_lds(const __bf16* gsrc, __bf16* lds_dst) {
#ifdef HAVE_ASYNC_LDS
  __builtin_amdgcn_global_load_async_to_lds_b128(
      (gas_v4i*)(void*)gsrc, (las_v4i*)(void*)lds_dst, 0, 0);
#else
  *(uint4*)lds_dst = *(const uint4*)gsrc;
#endif
}

__device__ __forceinline__ void wait_async_tail4(void) {
#ifdef HAVE_ASYNC_LDS
  __builtin_amdgcn_s_wait_asynccnt(4);   // retire all but the 4 in-flight copies
#endif
}
__device__ __forceinline__ void wait_async_all(void) {
#ifdef HAVE_ASYNC_LDS
  __builtin_amdgcn_s_wait_asynccnt(0);
#endif
}

__device__ __forceinline__ float silu(float v) {
  return v / (1.0f + __expf(-v));
}

// ---------------------------------------------------------------------------
// K0a: plain fp32 -> bf16 conversion (activations)
// ---------------------------------------------------------------------------
__global__ __launch_bounds__(256)
void f32_to_bf16_kernel(const float* __restrict__ in, __bf16* __restrict__ out, int n) {
  const int i = blockIdx.x * 256 + threadIdx.x;
  if (i < n) out[i] = (__bf16)in[i];
}

// ---------------------------------------------------------------------------
// K0b: transposing fp32 -> bf16 conversion (weights). in: [R][Ccols] fp32,
// out: [Ccols][R] bf16. Makes the GEMM B operand k-contiguous so B fragments
// load as ds_load_b128 (same as A) and async b128 copies need no transpose.
// ---------------------------------------------------------------------------
__global__ __launch_bounds__(256)
void f32_to_bf16_tr_kernel(const float* __restrict__ in, __bf16* __restrict__ out,
                           int R, int Ccols) {
  const int i = blockIdx.x * 256 + threadIdx.x;
  if (i < R * Ccols) {
    const int r = i / Ccols, c = i - r * Ccols;
    out[(size_t)c * R + r] = (__bf16)in[i];
  }
}

// ---------------------------------------------------------------------------
// WMMA GEMM: C[M][N] = A[M][K] @ B[K][N] + bias[N]; A bf16 [M][K] row-major,
// B supplied TRANSPOSED as BT bf16 [N][K]. C fp32.
// Block tile 128x128, K-step 32, 256 threads = 8 waves (4 M x 2 N), each wave
// a 32x64 sub-tile = 2x4 v_wmma_f32_16x16x32_bf16. Double-buffered LDS fed by
// GLOBAL_LOAD_ASYNC_TO_LDS_B128 (4 x 16B chunks per thread per tile).
// ---------------------------------------------------------------------------
#define BM 128
#define BN 128
#define BK 32
#define SK 48      // LDS row stride (bf16): 96B = 16B-aligned rows, bank-skewed

__global__ __launch_bounds__(256)
void gemm_bias_wmma_bf16(const __bf16* __restrict__ A, const __bf16* __restrict__ BT,
                         const float* __restrict__ bias, float* __restrict__ C,
                         int M, int N, int K)
{
  __shared__ __bf16 sA[2][BM][SK];    // (m, k)
  __shared__ __bf16 sB[2][BN][SK];    // (n, k)  -- BT is already n-major

  const int tid   = threadIdx.x;
  const int lane  = tid & 31;
  const int wave  = tid >> 5;
  const int waveM = wave & 3;         // 4 waves along M, 32 rows each
  const int waveN = wave >> 2;        // 2 waves along N, 64 cols each
  const int half  = lane >> 4;
  const int l16   = lane & 15;

  const int m0 = blockIdx.y * BM;
  const int n0 = blockIdx.x * BN;
  const int nk = K / BK;

  // Stage one 128x32 A tile + 128x32 BT tile: 512 + 512 16B chunks, 4/thread.
  auto stage = [&](int kt, int buf) {
    #pragma unroll
    for (int t = 0; t < 2; t++) {
      const int c   = tid + t * 256;           // chunk id, 0..511
      const int row = c >> 2;                  // 4 chunks per 32-wide row
      const int kk  = (c & 3) << 3;
      copy16_to_lds(A + (size_t)(m0 + row) * K + kt * BK + kk, &sA[buf][row][kk]);
    }
    #pragma unroll
    for (int t = 0; t < 2; t++) {
      const int c   = tid + t * 256;
      const int row = c >> 2;
      const int kk  = (c & 3) << 3;
      copy16_to_lds(BT + (size_t)(n0 + row) * K + kt * BK + kk, &sB[buf][row][kk]);
    }
  };

  v8f acc[2][4];
  #pragma unroll
  for (int i = 0; i < 2; i++)
    #pragma unroll
    for (int j = 0; j < 4; j++) { v8f z = {}; acc[i][j] = z; }

  stage(0, 0);

  for (int kt = 0; kt < nk; kt++) {
    const int buf = kt & 1;
    const bool more = (kt + 1 < nk);
    if (more) stage(kt + 1, buf ^ 1);          // tile kt+1 streams into other buffer
    if (more) wait_async_tail4(); else wait_async_all();   // tile kt's copies done
    __syncthreads();

    // Fragments per ISA 16-bit layouts; both operands k-contiguous -> b128 loads.
    v16bf afrag[2], bfrag[4];
    #pragma unroll
    for (int i = 0; i < 2; i++) {
      const int m = waveM * 32 + i * 16 + l16;
      #pragma unroll
      for (int e = 0; e < 16; e++) {
        // A 16x32: VGPR0..3 K=(0..7)+8*half, VGPR4..7 K=(16..23)+8*half
        const int k = e + 8 * half + ((e >= 8) ? 8 : 0);
        afrag[i][e] = sA[buf][m][k];
      }
    }
    #pragma unroll
    for (int j = 0; j < 4; j++) {
      const int n = waveN * 64 + j * 16 + l16;
      #pragma unroll
      for (int e = 0; e < 16; e++) {
        // B 32x16: VGPR0..3 K=(0..7)+8*half, VGPR4..7 K=(16..23)+8*half
        const int k = e + 8 * half + ((e >= 8) ? 8 : 0);
        bfrag[j][e] = sB[buf][n][k];
      }
    }

    #pragma unroll
    for (int i = 0; i < 2; i++)
      #pragma unroll
      for (int j = 0; j < 4; j++)
        acc[i][j] = __builtin_amdgcn_wmma_f32_16x16x32_bf16(
            false, afrag[i], false, bfrag[j], (short)0, acc[i][j], false, false);

    __syncthreads();   // all waves done with buf before tile kt+2 overwrites it
  }

  // Epilogue: fused bias. D layout: M = v + 8*half, N = lane&15.
  #pragma unroll
  for (int j = 0; j < 4; j++) {
    const int n  = n0 + waveN * 64 + j * 16 + l16;
    const float bv = bias[n];
    #pragma unroll
    for (int i = 0; i < 2; i++) {
      #pragma unroll
      for (int v = 0; v < 8; v++) {
        const int m = m0 + waveM * 32 + i * 16 + v + 8 * half;
        C[(size_t)m * N + n] = acc[i][j][v] + bv;
      }
    }
  }
}

// ---------------------------------------------------------------------------
// K2: causal depthwise conv(4) + SiLU on x_inner, SiLU on res gate
// ---------------------------------------------------------------------------
__global__ __launch_bounds__(256)
void conv_silu_kernel(const float* __restrict__ xr,
                      const float* __restrict__ conv_w,
                      const float* __restrict__ conv_b,
                      float* __restrict__ xc, float* __restrict__ rs)
{
  const int idx = blockIdx.x * blockDim.x + threadIdx.x;   // over NROWS*DI
  if (idx >= NROWS * DI) return;
  const int d  = idx & (DI - 1);
  const int bl = idx >> 10;
  const int l  = bl & (LL - 1);

  float acc = conv_b[d];
  #pragma unroll
  for (int j = 0; j < DC; j++) {
    const int ll = l - (DC - 1) + j;
    if (ll >= 0)
      acc += conv_w[d * DC + j] * xr[(size_t)(bl - (l - ll)) * (2 * DI) + d];
  }
  xc[idx] = silu(acc);
  rs[idx] = silu(xr[(size_t)bl * (2 * DI) + DI + d]);
}

// ---------------------------------------------------------------------------
// K3: per-row skinny GEMM (33 cols) + softplus(dt) + Abar = exp(dt * -exp(A_log))
// Output per row: [B(16) | C(16) | Abar(16)] = 48 f32
// ---------------------------------------------------------------------------
__global__ __launch_bounds__(128)
void bcdt_kernel(const float* __restrict__ xc, const float* __restrict__ W_x,
                 const float* __restrict__ b_x, const float* __restrict__ A_log,
                 float* __restrict__ bc)
{
  const int row = blockIdx.x;
  __shared__ float sx[DI];
  __shared__ float sdt;

  for (int i = threadIdx.x; i < DI; i += blockDim.x)
    sx[i] = xc[(size_t)row * DI + i];
  __syncthreads();

  const int c = threadIdx.x;
  float acc = 0.0f;
  if (c < 2 * DS + 1) {
    acc = b_x[c];
    for (int k = 0; k < DI; k++)
      acc += sx[k] * W_x[k * (2 * DS + 1) + c];
  }
  if (c == 2 * DS) {
    sdt = (acc > 20.0f) ? acc : log1pf(__expf(acc));   // softplus
  }
  __syncthreads();

  if (c < DS) {
    bc[(size_t)row * 48 + c] = acc;                        // B_sel
    const float Aneg = -__expf(A_log[c]);
    bc[(size_t)row * 48 + 32 + c] = __expf(sdt * Aneg);    // Abar in (0,1)
  } else if (c < 2 * DS) {
    bc[(size_t)row * 48 + c] = acc;                        // C_sel
  }
}

// ---------------------------------------------------------------------------
// K4: selective scan, one thread per (b, d) channel; per-timestep shared
// params staged through LDS in 32-step tiles. Writes y directly as bf16.
// ---------------------------------------------------------------------------
#define TL 32
__global__ __launch_bounds__(256)
void scan_kernel(const float* __restrict__ bc, const float* __restrict__ xc,
                 const float* __restrict__ rs, const float* __restrict__ Dvec,
                 __bf16* __restrict__ y)
{
  const int d = blockIdx.x * 256 + threadIdx.x;
  const int b = blockIdx.y;
  __shared__ float sbc[TL][48];

  float h[DS];
  #pragma unroll
  for (int s = 0; s < DS; s++) h[s] = 0.0f;
  const float Dd = Dvec[d];

  for (int l0 = 0; l0 < LL; l0 += TL) {
    __syncthreads();
    for (int i = threadIdx.x; i < TL * 48; i += 256)
      sbc[i / 48][i % 48] = bc[((size_t)(b * LL + l0) * 48) + i];
    __syncthreads();

    #pragma unroll 4
    for (int t = 0; t < TL; t++) {
      const size_t row = (size_t)(b * LL + l0 + t);
      const float x = xc[row * DI + d];
      const float r = rs[row * DI + d];
      float accy = 0.0f;
      #pragma unroll
      for (int s = 0; s < DS; s++) {
        h[s] = sbc[t][32 + s] * h[s] + sbc[t][s] * x;   // h = Abar*h + B*x
        accy += sbc[t][16 + s] * h[s];                  // y += C*h
      }
      y[row * DI + d] = (__bf16)((accy + Dd * x) * r);
    }
  }
}

// ---------------------------------------------------------------------------
// Launcher
// ---------------------------------------------------------------------------
extern "C" void kernel_launch(void* const* d_in, const int* in_sizes, int n_in,
                              void* d_out, int out_size, void* d_ws, size_t ws_size,
                              hipStream_t stream) {
  (void)in_sizes; (void)n_in; (void)out_size; (void)ws_size;

  const float* x      = (const float*)d_in[0];   // [B,L,DM]
  const float* W_in   = (const float*)d_in[1];   // [DM, 2*DI]
  const float* b_in   = (const float*)d_in[2];   // [2*DI]
  const float* conv_w = (const float*)d_in[3];   // [DI,1,DC]
  const float* conv_b = (const float*)d_in[4];   // [DI]
  const float* W_x    = (const float*)d_in[5];   // [DI, 2*DS+1]
  const float* b_x    = (const float*)d_in[6];   // [2*DS+1]
  const float* A_log  = (const float*)d_in[7];   // [DS]
  const float* Dvec   = (const float*)d_in[8];   // [DI]
  const float* W_out  = (const float*)d_in[9];   // [DI, DM]
  const float* b_out  = (const float*)d_in[10];  // [DM]
  float* out = (float*)d_out;                    // [B,L,DM]

  // Workspace layout (fp32 first, then bf16; whole set ~80MB -> lives in L2)
  float*  xr   = (float*)d_ws;                       // NROWS * 2*DI
  float*  xc   = xr + (size_t)NROWS * (2 * DI);      // NROWS * DI
  float*  rs   = xc + (size_t)NROWS * DI;            // NROWS * DI
  float*  bc   = rs + (size_t)NROWS * DI;            // NROWS * 48
  __bf16* xbf  = (__bf16*)(bc + (size_t)NROWS * 48); // NROWS * DM
  __bf16* w1bf = xbf + (size_t)NROWS * DM;           // [2*DI][DM]  (W_in^T)
  __bf16* w2bf = w1bf + (size_t)DM * (2 * DI);       // [DM][DI]    (W_out^T)
  __bf16* ybf  = w2bf + (size_t)DI * DM;             // NROWS * DI

  // K0: one-time bf16 conversion; weights transposed to n-major for the GEMM B path
  f32_to_bf16_kernel<<<(NROWS * DM) / 256, 256, 0, stream>>>(x, xbf, NROWS * DM);
  f32_to_bf16_tr_kernel<<<(DM * 2 * DI) / 256, 256, 0, stream>>>(W_in, w1bf, DM, 2 * DI);
  f32_to_bf16_tr_kernel<<<(DI * DM) / 256, 256, 0, stream>>>(W_out, w2bf, DI, DM);

  // K1: xr = x @ W_in + b_in    (4096 x 2048 x 512)
  gemm_bias_wmma_bf16<<<dim3((2 * DI) / BN, NROWS / BM), 256, 0, stream>>>(
      xbf, w1bf, b_in, xr, NROWS, 2 * DI, DM);

  // K2: conv + SiLU + gate SiLU
  conv_silu_kernel<<<(NROWS * DI) / 256, 256, 0, stream>>>(
      xr, conv_w, conv_b, xc, rs);

  // K3: B/C/dt projection + Abar precompute
  bcdt_kernel<<<NROWS, 128, 0, stream>>>(xc, W_x, b_x, A_log, bc);

  // K4: channel-parallel selective scan (writes bf16)
  scan_kernel<<<dim3(DI / 256, BB), 256, 0, stream>>>(bc, xc, rs, Dvec, ybf);

  // K5: out = y @ W_out + b_out  (4096 x 512 x 1024)
  gemm_bias_wmma_bf16<<<dim3(DM / BN, NROWS / BM), 256, 0, stream>>>(
      ybf, w2bf, b_out, out, NROWS, DM, DI);
}
